// A2MTransformer_90159953478029
// MI455X (gfx1250) — matrix operand order repaired and checked by reading, not compiled, not run on hardware
//
#include <hip/hip_runtime.h>

// ---------------------------------------------------------------------------
// CDNA5 (gfx1250) wave32 WMMA implementation of the conv+transformer model.
// All GEMMs use v_wmma_f32_16x16x32_f16 (f16 staged in LDS, f32 accumulate).
// Attention exploits the rel-position mask: it is exactly banded (|k-q|<=99),
// so each 16-query tile needs only a 256-wide key window held in LDS.
// Staging uses clamp+select (no predicated loads) and float4 vector loads.
// ---------------------------------------------------------------------------

typedef __attribute__((ext_vector_type(16))) _Float16 v16h;
typedef __attribute__((ext_vector_type(8)))  _Float16 v8h;
typedef __attribute__((ext_vector_type(4)))  _Float16 v4h;
typedef __attribute__((ext_vector_type(8)))  float    v8f;
typedef __attribute__((ext_vector_type(4)))  float    v4f;

__device__ __forceinline__ v16h mk16(v8h lo, v8h hi) {
    v16h r;
#pragma unroll
    for (int i = 0; i < 8; ++i) { r[i] = lo[i]; r[i + 8] = hi[i]; }
    return r;
}

__device__ __forceinline__ v4h cvt4(v4f v) {
    v4h h;
#pragma unroll
    for (int j = 0; j < 4; ++j) h[j] = (_Float16)v[j];
    return h;
}

// ---------------------------------------------------------------------------
// Generic GEMM: C[M,N] = alpha * A[M,K] * B(K,N)  (+bias)(+C if beta)(relu)
//  - A fp32 row-major; row index shifted by `ashift`, zero outside [0,arows)
//  - transB=1: B is (N,K) row-major, element stride `bstride`, offset boff
//  - transB=0: B is (K,N) row-major
// 256 threads = 8 waves (4x2), block tile 128x128, wave tile 32x64, K-step 32.
// ---------------------------------------------------------------------------
__global__ __launch_bounds__(256) void gemm_f16_wmma(
    const float* __restrict__ A, const float* __restrict__ B,
    float* __restrict__ C, const float* __restrict__ bias,
    int M, int N, int K, int lda, int ldb, int ldc,
    int transB, int bstride, int boff,
    int ashift, int arows, float alpha, int beta, int relu)
{
    __shared__ _Float16 As[128 * 40];   // [m][k], stride 40 halves (16B-aligned rows)
    __shared__ _Float16 Bs[128 * 40];   // [n][k]

    const int tid  = threadIdx.x;
    const int lane = tid & 31;
    const int wave = tid >> 5;
    const int wm = wave >> 1, wn = wave & 1;        // 4x2 wave grid
    const int ml = lane & 15, hh = lane >> 4;
    const int bm = blockIdx.y * 128, bn = blockIdx.x * 128;

    const bool afast  = (bm + ashift >= 0) && (bm + 127 + ashift < arows);
    const bool bfastT = transB && (bstride == 1) && (bn + 128 <= N);
    const bool bfastN = (!transB) && (bn + 128 <= N);

    // A / B-trans staging coords: 8 threads per 32-wide row, 32 rows per pass
    const int srow  = tid >> 3;           // 0..31
    const int scol4 = (tid & 7) * 4;      // 0,4,..,28
    // B non-trans staging coords: 32 threads per 128-wide row, 8 rows per pass
    const int bk_t = tid >> 5;            // 0..7
    const int bn4  = (tid & 31) * 4;      // 0..124

    v8f acc[2][4] = {};

    for (int k0 = 0; k0 < K; k0 += 32) {
        const int kmax = K - k0;          // multiple of 4 in this model
        __syncthreads();

        // ---------------- stage A tile (128 x 32) ----------------
        if (afast && kmax >= 32) {
#pragma unroll
            for (int it = 0; it < 4; ++it) {
                int i = it * 32 + srow;
                const float* src = A + (size_t)(bm + i + ashift) * lda + k0 + scol4;
                __builtin_prefetch(src + 32);          // next K-slab
                *(v4h*)&As[i * 40 + scol4] = cvt4(*(const v4f*)src);
            }
        } else {
#pragma unroll
            for (int it = 0; it < 4; ++it) {
                int i = it * 32 + srow;
                int ar = bm + i + ashift;
                int arc = min(max(ar, 0), arows - 1);
                bool ok = (ar >= 0) && (ar < arows) && (scol4 < kmax);
                int c4 = ok ? scol4 : 0;
                v4f v = *(const v4f*)(A + (size_t)arc * lda + k0 + c4);
                v4h hv;
#pragma unroll
                for (int j = 0; j < 4; ++j) hv[j] = ok ? (_Float16)v[j] : (_Float16)0.f;
                *(v4h*)&As[i * 40 + scol4] = hv;
            }
        }

        // ---------------- stage B tile into [n][k] ----------------
        if (bfastT && kmax >= 32) {
#pragma unroll
            for (int it = 0; it < 4; ++it) {
                int n = it * 32 + srow;
                const float* src = B + (size_t)(bn + n) * ldb + k0 + scol4 + boff;
                __builtin_prefetch(src + 32);
                *(v4h*)&Bs[n * 40 + scol4] = cvt4(*(const v4f*)src);
            }
        } else if (transB) {
#pragma unroll
            for (int it = 0; it < 4; ++it) {
                int n = it * 32 + srow;
                int gn = bn + n;
                int gnc = min(gn, N - 1);
                bool nv = gn < N;
                v4h hv;
#pragma unroll
                for (int j = 0; j < 4; ++j) {
                    int gk = k0 + scol4 + j;
                    int gkc = min(gk, K - 1);
                    float v = B[(size_t)gnc * ldb + (size_t)gkc * bstride + boff];
                    hv[j] = (nv && gk < K) ? (_Float16)v : (_Float16)0.f;
                }
                *(v4h*)&Bs[n * 40 + scol4] = hv;
            }
        } else if (bfastN && kmax >= 32) {
#pragma unroll
            for (int it = 0; it < 4; ++it) {
                int k = it * 8 + bk_t;
                const float* src = B + (size_t)(k0 + k) * ldb + bn + bn4;
                __builtin_prefetch(src + (size_t)32 * ldb);
                v4f v = *(const v4f*)src;
#pragma unroll
                for (int j = 0; j < 4; ++j)
                    Bs[(bn4 + j) * 40 + k] = (_Float16)v[j];
            }
        } else {
#pragma unroll
            for (int it = 0; it < 4; ++it) {
                int k = it * 8 + bk_t;
                int gk = k0 + k;
                int gkc = min(gk, K - 1);
                bool kv = gk < K;
#pragma unroll
                for (int j = 0; j < 4; ++j) {
                    int gn = bn + bn4 + j;
                    int gnc = min(gn, N - 1);
                    float v = B[(size_t)gkc * ldb + gnc];
                    Bs[(bn4 + j) * 40 + k] = (kv && gn < N) ? (_Float16)v : (_Float16)0.f;
                }
            }
        }
        __syncthreads();

        // ---------------- fragments + WMMA ----------------
        v16h af[2];
#pragma unroll
        for (int fm = 0; fm < 2; ++fm) {
            const _Float16* p = &As[(wm * 32 + fm * 16 + ml) * 40 + hh * 16];
            af[fm] = mk16(*(const v8h*)p, *(const v8h*)(p + 8));
        }
#pragma unroll
        for (int fn = 0; fn < 4; ++fn) {
            const _Float16* p = &Bs[(wn * 64 + fn * 16 + ml) * 40 + hh * 16];
            v16h bf = mk16(*(const v8h*)p, *(const v8h*)(p + 8));
#pragma unroll
            for (int fm = 0; fm < 2; ++fm)
                acc[fm][fn] = __builtin_amdgcn_wmma_f32_16x16x32_f16(
                    false, af[fm], false, bf, (short)0, acc[fm][fn], false, false);
        }
    }

    // ---------------- epilogue ----------------
#pragma unroll
    for (int fm = 0; fm < 2; ++fm)
#pragma unroll
        for (int fn = 0; fn < 4; ++fn)
#pragma unroll
            for (int r = 0; r < 8; ++r) {
                int gr = bm + wm * 32 + fm * 16 + r + 8 * hh;
                int gc = bn + wn * 64 + fn * 16 + ml;
                if (gr < M && gc < N) {
                    float v = acc[fm][fn][r] * alpha;
                    if (bias) v += bias[gc];
                    if (beta) v += C[(size_t)gr * ldc + gc];
                    if (relu) v = fmaxf(v, 0.f);
                    C[(size_t)gr * ldc + gc] = v;
                }
            }
}

// ---------------------------------------------------------------------------
// Banded relative attention. Q/K/V/O layout: (B*T, 768) head-concat fp32
// (head h occupies columns [h*96, h*96+96)). One wave per (b, h, 16-q tile).
// ---------------------------------------------------------------------------
__global__ __launch_bounds__(32) void attn_banded_wmma(
    const float* __restrict__ Qg, const float* __restrict__ Kg,
    const float* __restrict__ Vg, const float* __restrict__ relg,
    float* __restrict__ Og)
{
    __shared__ _Float16 Qs[16 * 96];
    __shared__ _Float16 Ks[64 * 96];    // B-layout [n][k] staging chunk
    __shared__ float    Sf[16 * 256];
    __shared__ float    Rf[16 * 200];   // rel term, indexed by (k-q+99)
    __shared__ _Float16 Ps[16 * 256];
    __shared__ _Float16 Vt[96 * 64];    // V^T chunk

    const int T = 1024, NREL = 199;
    const int q0 = blockIdx.x * 16;
    const int h = blockIdx.y, b = blockIdx.z;
    const size_t rowbase = (size_t)b * T;
    const float* Qh = Qg + rowbase * 768 + h * 96;
    const float* Kh = Kg + rowbase * 768 + h * 96;
    const float* Vh = Vg + rowbase * 768 + h * 96;
    const float* relh = relg + (size_t)h * NREL * 96;
    float* Oh = Og + rowbase * 768 + h * 96;

    const int lane = threadIdx.x;
    const int ml = lane & 15, hh = lane >> 4;
    const int kstart = q0 - 112;    // window [q0-112, q0+143] covers band +-99

    // ---- stage Q tile (16x96) as f16, vectorized ----
#pragma unroll
    for (int it = 0; it < 12; ++it) {
        int idx = it * 32 + lane;
        int row = idx / 24, c4 = (idx % 24) * 4;
        v4f v = *(const v4f*)(Qh + (size_t)(q0 + row) * 768 + c4);
        *(v4h*)&Qs[row * 96 + c4] = cvt4(v);
    }
    __syncthreads();

    // ---- R = Q @ rel_emb^T  (16 x 199) ----
    for (int c = 0; c < 4; ++c) {
#pragma unroll
        for (int it = 0; it < 48; ++it) {
            int idx = it * 32 + lane;
            int n = idx / 24, c4 = (idx % 24) * 4;
            int gn = c * 64 + n;
            int gnc = min(gn, NREL - 1);
            bool ok = gn < NREL;
            v4f v = *(const v4f*)(relh + (size_t)gnc * 96 + c4);
            v4h hv;
#pragma unroll
            for (int j = 0; j < 4; ++j) hv[j] = ok ? (_Float16)v[j] : (_Float16)0.f;
            *(v4h*)&Ks[n * 96 + c4] = hv;
        }
        __syncthreads();
        v8f acc[4] = {};
        for (int kc = 0; kc < 3; ++kc) {
            const _Float16* pa = &Qs[ml * 96 + kc * 32 + hh * 16];
            v16h a = mk16(*(const v8h*)pa, *(const v8h*)(pa + 8));
#pragma unroll
            for (int fn = 0; fn < 4; ++fn) {
                const _Float16* pb = &Ks[(fn * 16 + ml) * 96 + kc * 32 + hh * 16];
                v16h bb = mk16(*(const v8h*)pb, *(const v8h*)(pb + 8));
                acc[fn] = __builtin_amdgcn_wmma_f32_16x16x32_f16(
                    false, a, false, bb, (short)0, acc[fn], false, false);
            }
        }
#pragma unroll
        for (int fn = 0; fn < 4; ++fn)
#pragma unroll
            for (int r = 0; r < 8; ++r) {
                int m = r + 8 * hh, n = c * 64 + fn * 16 + ml;
                if (n < NREL) Rf[m * 200 + n] = acc[fn][r];
            }
        __syncthreads();
    }

    // ---- S = (Q@K^T)/sqrt(96) + rel, banded mask ----
    const float scale = 0.10206207262f;  // 1/sqrt(96)
    for (int c = 0; c < 4; ++c) {
#pragma unroll
        for (int it = 0; it < 48; ++it) {
            int idx = it * 32 + lane;
            int n = idx / 24, c4 = (idx % 24) * 4;
            int gk = kstart + c * 64 + n;
            int gkc = min(max(gk, 0), T - 1);
            bool ok = (gk >= 0) && (gk < T);
            v4f v = *(const v4f*)(Kh + (size_t)gkc * 768 + c4);
            v4h hv;
#pragma unroll
            for (int j = 0; j < 4; ++j) hv[j] = ok ? (_Float16)v[j] : (_Float16)0.f;
            *(v4h*)&Ks[n * 96 + c4] = hv;
        }
        __syncthreads();
        v8f acc[4] = {};
        for (int kc = 0; kc < 3; ++kc) {
            const _Float16* pa = &Qs[ml * 96 + kc * 32 + hh * 16];
            v16h a = mk16(*(const v8h*)pa, *(const v8h*)(pa + 8));
#pragma unroll
            for (int fn = 0; fn < 4; ++fn) {
                const _Float16* pb = &Ks[(fn * 16 + ml) * 96 + kc * 32 + hh * 16];
                v16h bb = mk16(*(const v8h*)pb, *(const v8h*)(pb + 8));
                acc[fn] = __builtin_amdgcn_wmma_f32_16x16x32_f16(
                    false, a, false, bb, (short)0, acc[fn], false, false);
            }
        }
#pragma unroll
        for (int fn = 0; fn < 4; ++fn)
#pragma unroll
            for (int r = 0; r < 8; ++r) {
                int m = r + 8 * hh;
                int n = c * 64 + fn * 16 + ml;
                int gk = kstart + n;
                int dd = gk - (q0 + m) + 99;
                float v = -1.0e8f;
                if (gk >= 0 && gk < T && dd >= 0 && dd < NREL)
                    v = acc[fn][r] * scale + Rf[m * 200 + dd];
                Sf[m * 256 + n] = v;
            }
        __syncthreads();
    }

    // ---- softmax over window (2 lanes per row, exp computed once) ----
    {
        int m = ml, st = hh;
        float mx = -3.0e38f;
        for (int j = st; j < 256; j += 2) mx = fmaxf(mx, Sf[m * 256 + j]);
        mx = fmaxf(mx, __shfl_xor(mx, 16, 32));
        float sum = 0.f;
        for (int j = st; j < 256; j += 2) {
            float e = expf(Sf[m * 256 + j] - mx);
            Sf[m * 256 + j] = e;
            sum += e;
        }
        sum += __shfl_xor(sum, 16, 32);
        float inv = 1.f / sum;
        for (int j = st; j < 256; j += 2)
            Ps[m * 256 + j] = (_Float16)(Sf[m * 256 + j] * inv);
    }
    __syncthreads();

    // ---- O = P @ V  (16 x 96) ----
    v8f oacc[6] = {};
    for (int c = 0; c < 4; ++c) {
#pragma unroll
        for (int it = 0; it < 48; ++it) {
            int idx = it * 32 + lane;
            int n = idx / 24, d4 = (idx % 24) * 4;
            int gk = kstart + c * 64 + n;
            int gkc = min(max(gk, 0), T - 1);
            bool ok = (gk >= 0) && (gk < T);
            v4f v = *(const v4f*)(Vh + (size_t)gkc * 768 + d4);
#pragma unroll
            for (int j = 0; j < 4; ++j)
                Vt[(d4 + j) * 64 + n] = ok ? (_Float16)v[j] : (_Float16)0.f;
        }
        __syncthreads();
#pragma unroll
        for (int kc = 0; kc < 2; ++kc) {
            const _Float16* pa = &Ps[ml * 256 + c * 64 + kc * 32 + hh * 16];
            v16h a = mk16(*(const v8h*)pa, *(const v8h*)(pa + 8));
#pragma unroll
            for (int fn = 0; fn < 6; ++fn) {
                const _Float16* pb = &Vt[(fn * 16 + ml) * 64 + kc * 32 + hh * 16];
                v16h bb = mk16(*(const v8h*)pb, *(const v8h*)(pb + 8));
                oacc[fn] = __builtin_amdgcn_wmma_f32_16x16x32_f16(
                    false, a, false, bb, (short)0, oacc[fn], false, false);
            }
        }
        __syncthreads();
    }
#pragma unroll
    for (int fn = 0; fn < 6; ++fn)
#pragma unroll
        for (int r = 0; r < 8; ++r) {
            int m = r + 8 * hh, d = fn * 16 + ml;
            Oh[(size_t)(q0 + m) * 768 + d] = oacc[fn][r];
        }
}

// ---------------------------------------------------------------------------
// Small helper kernels
// ---------------------------------------------------------------------------
__global__ void k_fold_bn(const float* __restrict__ g, const float* __restrict__ be,
                          const float* __restrict__ mn, const float* __restrict__ vr,
                          const float* __restrict__ cb, float* __restrict__ s,
                          float* __restrict__ t, int C)
{
    int i = blockIdx.x * 256 + threadIdx.x;
    if (i >= C) return;
    float sc = g[i] * rsqrtf(vr[i] + 1e-5f);
    s[i] = sc;
    t[i] = be[i] - mn[i] * sc + cb[i] * sc;
}

__global__ void k_scale_shift(float* __restrict__ y, const float* __restrict__ x,
                              const float* __restrict__ res, const float* __restrict__ s,
                              const float* __restrict__ t, int n, int C, int relu)
{
    int i = blockIdx.x * 256 + threadIdx.x;
    if (i >= n) return;
    int c = i % C;
    float v = x[i] * s[c] + t[c];
    if (res) v += res[i];
    if (relu) v = fmaxf(v, 0.f);
    y[i] = v;
}

__global__ __launch_bounds__(256) void k_add_ln(
    const float* __restrict__ x, const float* __restrict__ r,
    const float* __restrict__ g, const float* __restrict__ be,
    float* __restrict__ o, int D)
{
    __shared__ float red[256];
    const int row = blockIdx.x, tid = threadIdx.x;
    const float* xr = x + (size_t)row * D;
    const float* rr = r + (size_t)row * D;
    float s = 0.f;
    for (int c = tid; c < D; c += 256) s += xr[c] + rr[c];
    red[tid] = s; __syncthreads();
    for (int st = 128; st > 0; st >>= 1) { if (tid < st) red[tid] += red[tid + st]; __syncthreads(); }
    const float mean = red[0] * (1.f / D);
    __syncthreads();
    float vs = 0.f;
    for (int c = tid; c < D; c += 256) { float d = xr[c] + rr[c] - mean; vs += d * d; }
    red[tid] = vs; __syncthreads();
    for (int st = 128; st > 0; st >>= 1) { if (tid < st) red[tid] += red[tid + st]; __syncthreads(); }
    const float inv = rsqrtf(red[0] * (1.f / D) + 1e-5f);
    float* orow = o + (size_t)row * D;
    for (int c = tid; c < D; c += 256) orow[c] = (xr[c] + rr[c] - mean) * inv * g[c] + be[c];
}

__global__ void k_transpose_in(const float* __restrict__ x, float* __restrict__ X0)
{
    int i = blockIdx.x * 256 + threadIdx.x;          // over B*CIN*L
    if (i >= 4 * 8 * 1024) return;
    int t = i % 1024, c = (i / 1024) % 8, b = i / (1024 * 8);
    X0[((size_t)b * 1024 + t) * 8 + c] = x[i];
}

__global__ void k_transpose_out(const float* __restrict__ O, float* __restrict__ out)
{
    int i = blockIdx.x * 256 + threadIdx.x;          // over B*COUT*L
    if (i >= 4 * 80 * 1024) return;
    int t = i % 1024, c = (i / 1024) % 80, b = i / (1024 * 80);
    out[i] = O[((size_t)b * 1024 + t) * 80 + c];
}

// wq/wk/wv (H, D, DQ) -> head-concat (D, H*DQ)
__global__ void k_headcat(const float* __restrict__ w, float* __restrict__ o)
{
    int i = blockIdx.x * 256 + threadIdx.x;
    if (i >= 8 * 768 * 96) return;
    int h = i / (768 * 96);
    int f = (i / 96) % 768;
    int a = i % 96;
    o[(size_t)f * 768 + h * 96 + a] = w[i];
}

// ---------------------------------------------------------------------------
// Host orchestration
// ---------------------------------------------------------------------------
static inline void launch_gemm(hipStream_t st, const float* A, const float* Bm, float* Cm,
                               const float* bias, int M, int N, int K, int lda, int ldb, int ldc,
                               int transB, int bstride, int boff, int ashift, int arows,
                               float alpha, int beta, int relu)
{
    dim3 grid((N + 127) / 128, (M + 127) / 128);
    gemm_f16_wmma<<<grid, dim3(256), 0, st>>>(A, Bm, Cm, bias, M, N, K, lda, ldb, ldc,
                                              transB, bstride, boff, ashift, arows,
                                              alpha, beta, relu);
}

extern "C" void kernel_launch(void* const* d_in, const int* in_sizes, int n_in,
                              void* d_out, int out_size, void* d_ws, size_t ws_size,
                              hipStream_t stream)
{
    (void)in_sizes; (void)n_in; (void)out_size; (void)ws_size;

    int idx = 0;
    auto nxt = [&]() { return (const float*)d_in[idx++]; };

    const float* xin = nxt();
    struct BlkP { const float *c1w,*c1b,*bn1g,*bn1b,*bn1m,*bn1v,
                              *c2w,*c2b,*bn2g,*bn2b,*bn2m,*bn2v,
                              *rw,*rb,*rbng,*rbnb,*rbnm,*rbnv; };
    BlkP blk[3] = {};
    for (int i = 0; i < 3; ++i) {
        blk[i].c1w = nxt(); blk[i].c1b = nxt();
        blk[i].bn1g = nxt(); blk[i].bn1b = nxt(); blk[i].bn1m = nxt(); blk[i].bn1v = nxt();
        blk[i].c2w = nxt(); blk[i].c2b = nxt();
        blk[i].bn2g = nxt(); blk[i].bn2b = nxt(); blk[i].bn2m = nxt(); blk[i].bn2v = nxt();
        if (i == 0) {
            blk[i].rw = nxt(); blk[i].rb = nxt();
            blk[i].rbng = nxt(); blk[i].rbnb = nxt(); blk[i].rbnm = nxt(); blk[i].rbnv = nxt();
        }
    }
    const float* w_in_w = nxt(); const float* w_in_b = nxt();
    struct LyrP { const float *wq,*wk,*wv,*wo,*rel,*ln1g,*ln1b,*ln2g,*ln2b,*f1w,*f1b,*f2w,*f2b; };
    LyrP ly[6];
    for (int i = 0; i < 6; ++i) {
        ly[i].wq = nxt(); ly[i].wk = nxt(); ly[i].wv = nxt(); ly[i].wo = nxt();
        ly[i].rel = nxt();
        ly[i].ln1g = nxt(); ly[i].ln1b = nxt(); ly[i].ln2g = nxt(); ly[i].ln2b = nxt();
        ly[i].f1w = nxt(); ly[i].f1b = nxt(); ly[i].f2w = nxt(); ly[i].f2b = nxt();
    }
    const float* w_out_w = nxt(); const float* w_out_b = nxt();

    // ---- workspace carve (floats) ----
    float* ws = (float*)d_ws;
    size_t off = 0;
    auto alloc = [&](size_t n) { float* p = ws + off; off += n; return p; };
    float* ACT0 = alloc((size_t)4096 * 768);
    float* ACT1 = alloc((size_t)4096 * 768);
    float* BIG  = alloc((size_t)4096 * 3072);
    float* TMP  = alloc((size_t)4096 * 768);
    float* QB   = alloc((size_t)4096 * 768);   // head-concat (B*T, 768)
    float* KB   = alloc((size_t)4096 * 768);
    float* VB   = alloc((size_t)4096 * 768);
    float* OB   = alloc((size_t)4096 * 768);
    float* WQT  = alloc((size_t)768 * 768);    // repacked per-layer weights
    float* WKT  = alloc((size_t)768 * 768);
    float* WVT  = alloc((size_t)768 * 768);
    float* X0   = alloc((size_t)4096 * 8);
    float* SS   = alloc(4 * 768);
    float* OUTB = alloc((size_t)4096 * 80);
    float *S1 = SS, *T1v = SS + 768, *S2 = SS + 1536, *T2v = SS + 2304;

    // ---- input transpose: (B,CIN,L) -> rows (b*L+t, cin) ----
    k_transpose_in<<<(4 * 8 * 1024 + 255) / 256, 256, 0, stream>>>(xin, X0);

    auto conv3 = [&](const float* Xi, const float* W, float* Y, int Cin) {
        for (int b = 0; b < 4; ++b)
            for (int t = 0; t < 3; ++t)
                launch_gemm(stream, Xi + (size_t)b * 1024 * Cin, W,
                            Y + (size_t)b * 1024 * 768, nullptr,
                            1024, 768, Cin, Cin, Cin * 3, 768,
                            /*transB*/1, /*bstride*/3, /*boff*/t,
                            /*ashift*/t - 1, /*arows*/1024, 1.f, (t > 0), 0);
    };
    auto fold = [&](const float* g, const float* be, const float* mn, const float* vr,
                    const float* cb, float* s, float* t) {
        k_fold_bn<<<3, 256, 0, stream>>>(g, be, mn, vr, cb, s, t, 768);
    };
    auto ew = [&](float* y, const float* x, const float* res, const float* s,
                  const float* t, int relu) {
        int n = 4096 * 768;
        k_scale_shift<<<(n + 255) / 256, 256, 0, stream>>>(y, x, res, s, t, n, 768, relu);
    };

    auto run_block = [&](const float* Xi, int Cin, const BlkP& p, float* Xo) {
        conv3(Xi, p.c1w, BIG, Cin);
        fold(p.bn1g, p.bn1b, p.bn1m, p.bn1v, p.c1b, S1, T1v);
        ew(BIG, BIG, nullptr, S1, T1v, 1);
        conv3(BIG, p.c2w, TMP, 768);
        fold(p.bn2g, p.bn2b, p.bn2m, p.bn2v, p.c2b, S1, T1v);
        const float* res = Xi;
        if (p.rw) {
            for (int b = 0; b < 4; ++b)
                launch_gemm(stream, Xi + (size_t)b * 1024 * Cin, p.rw,
                            QB + (size_t)b * 1024 * 768, nullptr,
                            1024, 768, Cin, Cin, Cin, 768, 1, 1, 0, 0, 1024, 1.f, 0, 0);
            fold(p.rbng, p.rbnb, p.rbnm, p.rbnv, p.rb, S2, T2v);
            ew(QB, QB, nullptr, S2, T2v, 0);
            res = QB;
        }
        ew(Xo, TMP, res, S1, T1v, 1);
    };

    run_block(X0, 8, blk[0], ACT0);
    run_block(ACT0, 768, blk[1], ACT1);
    run_block(ACT1, 768, blk[2], ACT0);

    // ---- input projection ----
    launch_gemm(stream, ACT0, w_in_w, ACT1, w_in_b,
                4096, 768, 768, 768, 768, 768, 1, 1, 0, 0, 4096, 1.f, 0, 0);

    // ---- encoder layers (x lives in ACT1, ACT0 is LN scratch) ----
    float* X  = ACT1;
    float* XL = ACT0;
    const int HC = (8 * 768 * 96 + 255) / 256;
    for (int l = 0; l < 6; ++l) {
        const LyrP& p = ly[l];
        k_headcat<<<HC, 256, 0, stream>>>(p.wq, WQT);
        k_headcat<<<HC, 256, 0, stream>>>(p.wk, WKT);
        k_headcat<<<HC, 256, 0, stream>>>(p.wv, WVT);
        launch_gemm(stream, X, WQT, QB, nullptr, 4096, 768, 768, 768, 768, 768,
                    0, 1, 0, 0, 4096, 1.f, 0, 0);
        launch_gemm(stream, X, WKT, KB, nullptr, 4096, 768, 768, 768, 768, 768,
                    0, 1, 0, 0, 4096, 1.f, 0, 0);
        launch_gemm(stream, X, WVT, VB, nullptr, 4096, 768, 768, 768, 768, 768,
                    0, 1, 0, 0, 4096, 1.f, 0, 0);
        attn_banded_wmma<<<dim3(64, 8, 4), 32, 0, stream>>>(QB, KB, VB, p.rel, OB);
        // wo is contiguous as (H*DQ, D) row-major => single GEMM on concat-O
        launch_gemm(stream, OB, p.wo, TMP, nullptr, 4096, 768, 768, 768, 768, 768,
                    0, 1, 0, 0, 4096, 1.f, 0, 0);
        k_add_ln<<<4096, 256, 0, stream>>>(X, TMP, p.ln1g, p.ln1b, XL, 768);
        launch_gemm(stream, XL, p.f1w, BIG, p.f1b,
                    4096, 3072, 768, 768, 768, 3072, 1, 1, 0, 0, 4096, 1.f, 0, 1);
        launch_gemm(stream, BIG, p.f2w, TMP, p.f2b,
                    4096, 768, 3072, 3072, 3072, 768, 1, 1, 0, 0, 4096, 1.f, 0, 0);
        k_add_ln<<<4096, 256, 0, stream>>>(XL, TMP, p.ln2g, p.ln2b, X, 768);
    }

    // ---- output projection + transpose ----
    launch_gemm(stream, X, w_out_w, OUTB, w_out_b,
                4096, 80, 768, 768, 768, 80, 1, 1, 0, 0, 4096, 1.f, 0, 0);
    k_transpose_out<<<(4 * 80 * 1024 + 255) / 256, 256, 0, stream>>>(OUTB, (float*)d_out);
}
